// MultiHeadSelfAttention_6811818131464
// MI455X (gfx1250) — matrix operand order repaired and checked by reading, not compiled, not run on hardware
//
#include <hip/hip_runtime.h>
#include <hip/hip_bf16.h>
#include <math.h>

// ---------------------------------------------------------------------------
// MI455X (gfx1250) multi-head causal self-attention.
// B=2, S=2048, D=768, H=12, DK=64.  Compute-bound (~45 GFLOP vs ~35MB I/O):
// matmuls on v_wmma_f32_16x16x32_bf16 (fp32 accum). Attention K/V tiles are
// staged into LDS by the Tensor Data Mover (tensor_load_to_lds, TENSORcnt),
// double-buffered so DMA of block j+1 overlaps WMMA on block j. Softmax is
// register-resident (shfl_xor butterflies + VALU exp on accumulators).
// ---------------------------------------------------------------------------

typedef __bf16 bf16;
typedef __attribute__((ext_vector_type(16))) __bf16 v16bf;
typedef __attribute__((ext_vector_type(8)))  float   v8f;
typedef __attribute__((ext_vector_type(4)))  unsigned int v4u;
typedef __attribute__((ext_vector_type(8)))  unsigned int v8u;

union FragBF {
    v16bf v;
    uint4 u[2];
};

#define BB 2
#define SS 2048
#define DD 768
#define HH 12
#define DK 64

// ---------------- TDM: 2-D tile (dim1 rows x dim0 elems) -> LDS ------------
// Builds the D# (group0 + group1) in SGPRs and issues TENSOR_LOAD_TO_LDS.
// dims/strides in elements (data_size = 2 bytes, bf16); LDS rows packed.
__device__ __forceinline__ void tdm_load_2d(unsigned int lds_addr, const void* gaddr,
                                            unsigned int dim0, unsigned int dim1,
                                            unsigned int stride0) {
    unsigned long long ga = (unsigned long long)(uintptr_t)gaddr;
    v4u g0;
    g0[0] = 1u;                                   // count=1, user descriptor
    g0[1] = lds_addr;                             // LDS byte address
    g0[2] = (unsigned int)ga;                     // global_addr[31:0]
    g0[3] = (unsigned int)((ga >> 32) & 0x1FFFFFFu) | 0x80000000u; // addr[56:32] | type=2
    v8u g1;
    g1[0] = 1u << 16;                             // data_size = 1 (2 bytes)
    g1[1] = (dim0 & 0xFFFFu) << 16;               // tensor_dim0[15:0]
    g1[2] = (dim0 >> 16) | ((dim1 & 0xFFFFu) << 16);   // dim0[31:16] | dim1[15:0]
    g1[3] = (dim1 >> 16) | ((dim0 & 0xFFFFu) << 16);   // dim1[31:16] | tile_dim0
    g1[4] = (dim1 & 0xFFFFu);                     // tile_dim1 (tile == tensor), tile_dim2=0
    g1[5] = stride0;                              // tensor_dim0_stride[31:0]
    g1[6] = 0u;                                   // stride0[47:32] | stride1[15:0]
    g1[7] = 0u;
    asm volatile("tensor_load_to_lds %0, %1" :: "s"(g0), "s"(g1) : "memory");
}

// ------------------------------ fp32 -> bf16 -------------------------------
__global__ __launch_bounds__(256) void cvt_f32_bf16(const float* __restrict__ src,
                                                    bf16* __restrict__ dst, int n) {
    int i = blockIdx.x * 256 + threadIdx.x;
    if (i < n) dst[i] = (bf16)src[i];
}

// ------------------------- V transpose: (B,S,D)->(B,H,DK,S) ----------------
__global__ __launch_bounds__(256) void transpose_v(const bf16* __restrict__ V,
                                                   bf16* __restrict__ Vt) {
    int idx = blockIdx.x * 256 + threadIdx.x;          // over B*H*DK*S
    int s = idx & (SS - 1);
    int d = (idx >> 11) & (DK - 1);
    int h = (idx / (SS * DK)) % HH;
    int b = idx / (SS * DK * HH);
    Vt[idx] = V[((size_t)b * SS + s) * DD + h * DK + d];
}

// ------------------------------- NT GEMM -----------------------------------
// C[M,N] = A[M,K] (bf16, row-major) @ W[N,K]^T (bf16, row-major).
// Block: 256 threads = 8 waves, 4(M) x 2(N); wave tile 32x64, K-step 32.
template <bool F32OUT>
__global__ __launch_bounds__(256) void gemm_nt(const bf16* __restrict__ A,
                                               const bf16* __restrict__ W,
                                               void* __restrict__ Cout,
                                               int M, int N, int K) {
    const int lane = threadIdx.x & 31;
    const int wave = threadIdx.x >> 5;
    const int half = lane >> 4;
    const int l16  = lane & 15;

    const int mBase = blockIdx.x * 128 + (wave >> 1) * 32;
    const int nBase = blockIdx.y * 128 + (wave & 1) * 64;

    const bf16* apBase0 = A + (size_t)(mBase + l16) * K + half * 8;
    const bf16* apBase1 = A + (size_t)(mBase + 16 + l16) * K + half * 8;
    const bf16* bpBase[4];
#pragma unroll
    for (int nt = 0; nt < 4; ++nt)
        bpBase[nt] = W + (size_t)(nBase + nt * 16 + l16) * K + half * 8;

    v8f acc[2][4];
#pragma unroll
    for (int mt = 0; mt < 2; ++mt)
#pragma unroll
        for (int nt = 0; nt < 4; ++nt) acc[mt][nt] = (v8f)(0.0f);

    for (int k0 = 0; k0 < K; k0 += 32) {
        FragBF a[2];
        a[0].u[0] = *(const uint4*)(apBase0 + k0);
        a[0].u[1] = *(const uint4*)(apBase0 + k0 + 16);
        a[1].u[0] = *(const uint4*)(apBase1 + k0);
        a[1].u[1] = *(const uint4*)(apBase1 + k0 + 16);
        FragBF b[4];
#pragma unroll
        for (int nt = 0; nt < 4; ++nt) {
            b[nt].u[0] = *(const uint4*)(bpBase[nt] + k0);
            b[nt].u[1] = *(const uint4*)(bpBase[nt] + k0 + 16);
        }
        if (k0 + 32 < K) {
            __builtin_prefetch(apBase0 + k0 + 32, 0, 3);
            __builtin_prefetch(apBase1 + k0 + 32, 0, 3);
            __builtin_prefetch(bpBase[0] + k0 + 32, 0, 3);
            __builtin_prefetch(bpBase[2] + k0 + 32, 0, 3);
        }
#pragma unroll
        for (int mt = 0; mt < 2; ++mt)
#pragma unroll
            for (int nt = 0; nt < 4; ++nt)
                acc[mt][nt] = __builtin_amdgcn_wmma_f32_16x16x32_bf16(
                    false, a[mt].v, false, b[nt].v, (short)0, acc[mt][nt], false, false);
    }

#pragma unroll
    for (int mt = 0; mt < 2; ++mt)
#pragma unroll
        for (int nt = 0; nt < 4; ++nt)
#pragma unroll
            for (int r = 0; r < 8; ++r) {
                int row = mBase + mt * 16 + half * 8 + r;
                int col = nBase + nt * 16 + l16;
                if (F32OUT)
                    ((float*)Cout)[(size_t)row * N + col] = acc[mt][nt][r];
                else
                    ((bf16*)Cout)[(size_t)row * N + col] = (bf16)acc[mt][nt][r];
            }
}

// --------------------------- flash attention -------------------------------
// grid: (S/64, B*H); block: 128 threads = 4 waves; each wave owns 16 q-rows.
// K/V 64x64 tiles are shared by all 4 waves -> staged once per block by the
// TDM into double-buffered LDS; WMMA fragments come from ds_load_b128.
__global__ __launch_bounds__(128) void attn_kernel(const bf16* __restrict__ Q,
                                                   const bf16* __restrict__ Kmat,
                                                   const bf16* __restrict__ Vt,
                                                   bf16* __restrict__ Oc) {
    __shared__ __align__(16) bf16 sK[2][64][64];   // 8KB x2 : keys x dims
    __shared__ __align__(16) bf16 sV[2][64][64];   // 8KB x2 : dims x keys
    __shared__ __align__(16) bf16 sP[4][16][64];   // P transpose, 2KB/wave

    const int lane = threadIdx.x & 31;
    const int w    = threadIdx.x >> 5;
    const int half = lane >> 4;
    const int l16  = lane & 15;

    const int qb = blockIdx.x;                 // query block (64 rows)
    const int b  = blockIdx.y / HH;
    const int h  = blockIdx.y % HH;

    const float NEG_INF = -__builtin_inff();

    // Q fragments (K = DK = 64 -> two K-chunks of 32), loaded once from global.
    const int qrow = qb * 64 + w * 16 + l16;
    FragBF aQ[2];
#pragma unroll
    for (int kk = 0; kk < 2; ++kk) {
        const bf16* ap = Q + ((size_t)b * SS + qrow) * DD + h * DK + kk * 32 + half * 8;
        aQ[kk].u[0] = *(const uint4*)(ap);
        aQ[kk].u[1] = *(const uint4*)(ap + 16);
    }

    const bf16* Kg0 = Kmat + (size_t)b * SS * DD + h * DK;          // + key*DD
    const bf16* Vg0 = Vt + ((size_t)b * HH + h) * DK * SS;          // + d*SS + key

    // Prologue: DMA key-block 0 into buffer 0 (wave 0 only; TDM is per-wave).
    if (w == 0) {
        tdm_load_2d((unsigned int)(uintptr_t)&sK[0][0][0], Kg0, 64, 64, DD);
        tdm_load_2d((unsigned int)(uintptr_t)&sV[0][0][0], Vg0, 64, 64, SS);
    }

    v8f accO[4];
#pragma unroll
    for (int nt = 0; nt < 4; ++nt) accO[nt] = (v8f)(0.0f);

    float m[8], l[8];
#pragma unroll
    for (int r = 0; r < 8; ++r) { m[r] = NEG_INF; l[r] = 0.0f; }

    for (int j = 0; j <= qb; ++j) {
        const int buf = j & 1;
        if (w == 0) __builtin_amdgcn_s_wait_tensorcnt(0);   // tile j in LDS
        __syncthreads();
        if (w == 0 && j < qb) {                              // stream tile j+1
            tdm_load_2d((unsigned int)(uintptr_t)&sK[buf ^ 1][0][0],
                        Kg0 + (size_t)(j + 1) * 64 * DD, 64, 64, DD);
            tdm_load_2d((unsigned int)(uintptr_t)&sV[buf ^ 1][0][0],
                        Vg0 + (j + 1) * 64, 64, 64, SS);
        }

        // ---- S = Q @ K^T : fragments from LDS ----
        v8f accS[4];
#pragma unroll
        for (int nt = 0; nt < 4; ++nt) {
            FragBF bk0, bk1;
            const bf16* kp = &sK[buf][nt * 16 + l16][half * 8];
            bk0.u[0] = *(const uint4*)(kp);
            bk0.u[1] = *(const uint4*)(kp + 16);
            bk1.u[0] = *(const uint4*)(kp + 32);
            bk1.u[1] = *(const uint4*)(kp + 48);
            accS[nt] = __builtin_amdgcn_wmma_f32_16x16x32_bf16(
                false, aQ[0].v, false, bk0.v, (short)0, (v8f)(0.0f), false, false);
            accS[nt] = __builtin_amdgcn_wmma_f32_16x16x32_bf16(
                false, aQ[1].v, false, bk1.v, (short)0, accS[nt], false, false);
        }

        // ---- scale + causal mask (registers only) ----
#pragma unroll
        for (int nt = 0; nt < 4; ++nt)
#pragma unroll
            for (int r = 0; r < 8; ++r) accS[nt][r] *= 0.125f;   // 1/sqrt(DK)
        if (j == qb) {
            const int qg = qb * 64 + w * 16 + half * 8;
#pragma unroll
            for (int nt = 0; nt < 4; ++nt) {
                const int colg = j * 64 + nt * 16 + l16;
#pragma unroll
                for (int r = 0; r < 8; ++r)
                    if (colg > qg + r) accS[nt][r] = NEG_INF;
            }
        }

        // ---- row max via 16-lane butterfly ----
        float rmax[8];
#pragma unroll
        for (int r = 0; r < 8; ++r)
            rmax[r] = fmaxf(fmaxf(accS[0][r], accS[1][r]),
                            fmaxf(accS[2][r], accS[3][r]));
#pragma unroll
        for (int d = 1; d < 16; d <<= 1)
#pragma unroll
            for (int r = 0; r < 8; ++r)
                rmax[r] = fmaxf(rmax[r], __shfl_xor(rmax[r], d, 32));

        float mnew[8], alpha[8];
#pragma unroll
        for (int r = 0; r < 8; ++r) {
            mnew[r]  = fmaxf(m[r], rmax[r]);
            alpha[r] = __expf(m[r] - mnew[r]);
            m[r]     = mnew[r];
        }

        // ---- P = exp(S - m): VALU, bf16 transpose to LDS ----
        float rsum[8];
#pragma unroll
        for (int r = 0; r < 8; ++r) rsum[r] = 0.0f;
#pragma unroll
        for (int nt = 0; nt < 4; ++nt)
#pragma unroll
            for (int r = 0; r < 8; ++r) {
                float p = __expf(accS[nt][r] - mnew[r]);
                rsum[r] += p;
                sP[w][half * 8 + r][nt * 16 + l16] = (bf16)p;
            }
#pragma unroll
        for (int d = 1; d < 16; d <<= 1)
#pragma unroll
            for (int r = 0; r < 8; ++r)
                rsum[r] += __shfl_xor(rsum[r], d, 32);
#pragma unroll
        for (int r = 0; r < 8; ++r) l[r] = l[r] * alpha[r] + rsum[r];

        // ---- rescale O ----
#pragma unroll
        for (int nt = 0; nt < 4; ++nt)
#pragma unroll
            for (int r = 0; r < 8; ++r) accO[nt][r] *= alpha[r];

        __syncthreads();   // sP store (C-layout lanes) -> load (A-layout lanes)

        // ---- O += P @ V : P from sP, V fragments from sV ----
        FragBF aP[2];
#pragma unroll
        for (int kk = 0; kk < 2; ++kk) {
            const bf16* pp = &sP[w][l16][kk * 32 + half * 8];
            aP[kk].u[0] = *(const uint4*)(pp);
            aP[kk].u[1] = *(const uint4*)(pp + 16);
        }
#pragma unroll
        for (int nt = 0; nt < 4; ++nt) {
            FragBF bv0, bv1;
            const bf16* vp = &sV[buf][nt * 16 + l16][half * 8];
            bv0.u[0] = *(const uint4*)(vp);
            bv0.u[1] = *(const uint4*)(vp + 16);
            bv1.u[0] = *(const uint4*)(vp + 32);
            bv1.u[1] = *(const uint4*)(vp + 48);
            accO[nt] = __builtin_amdgcn_wmma_f32_16x16x32_bf16(
                false, aP[0].v, false, bv0.v, (short)0, accO[nt], false, false);
            accO[nt] = __builtin_amdgcn_wmma_f32_16x16x32_bf16(
                false, aP[1].v, false, bv1.v, (short)0, accO[nt], false, false);
        }
    }

    // ---- normalize and store ----
#pragma unroll
    for (int nt = 0; nt < 4; ++nt)
#pragma unroll
        for (int r = 0; r < 8; ++r) {
            int rl = half * 8 + r;
            float o = accO[nt][r] / l[r];
            int srow = qb * 64 + w * 16 + rl;
            Oc[((size_t)b * SS + srow) * DD + h * DK + nt * 16 + l16] = (bf16)o;
        }
}

// ------------------------------- launcher ----------------------------------
extern "C" void kernel_launch(void* const* d_in, const int* in_sizes, int n_in,
                              void* d_out, int out_size, void* d_ws, size_t ws_size,
                              hipStream_t stream) {
    const float* X  = (const float*)d_in[0];
    const float* Wq = (const float*)d_in[1];
    const float* Wk = (const float*)d_in[2];
    const float* Wv = (const float*)d_in[3];
    const float* Wo = (const float*)d_in[4];
    float* Out = (float*)d_out;

    const int M  = BB * SS;         // 4096
    const int nX = M * DD;          // 3145728
    const int nW = DD * DD;         // 589824

    char* base = (char*)d_ws;
    size_t off = 0;
    auto carve = [&](size_t bytes) {
        char* p = base + off;
        off += (bytes + 255) & ~(size_t)255;
        return p;
    };
    bf16* Xbf  = (bf16*)carve((size_t)nX * 2);
    bf16* Wqb  = (bf16*)carve((size_t)nW * 2);
    bf16* Wkb  = (bf16*)carve((size_t)nW * 2);
    bf16* Wvb  = (bf16*)carve((size_t)nW * 2);
    bf16* Wob  = (bf16*)carve((size_t)nW * 2);
    bf16* Qbf  = (bf16*)carve((size_t)nX * 2);
    bf16* Kbf  = (bf16*)carve((size_t)nX * 2);
    bf16* Vbf  = (bf16*)carve((size_t)nX * 2);
    bf16* Vtb  = (bf16*)carve((size_t)nX * 2);
    bf16* Attb = (bf16*)carve((size_t)nX * 2);

    // 1) fp32 -> bf16
    cvt_f32_bf16<<<(nX + 255) / 256, 256, 0, stream>>>(X,  Xbf, nX);
    cvt_f32_bf16<<<(nW + 255) / 256, 256, 0, stream>>>(Wq, Wqb, nW);
    cvt_f32_bf16<<<(nW + 255) / 256, 256, 0, stream>>>(Wk, Wkb, nW);
    cvt_f32_bf16<<<(nW + 255) / 256, 256, 0, stream>>>(Wv, Wvb, nW);
    cvt_f32_bf16<<<(nW + 255) / 256, 256, 0, stream>>>(Wo, Wob, nW);

    // 2) Q/K/V projections: [4096,768] @ [768,768]^T
    dim3 ggrid(M / 128, DD / 128);
    gemm_nt<false><<<ggrid, 256, 0, stream>>>(Xbf, Wqb, Qbf, M, DD, DD);
    gemm_nt<false><<<ggrid, 256, 0, stream>>>(Xbf, Wkb, Kbf, M, DD, DD);
    gemm_nt<false><<<ggrid, 256, 0, stream>>>(Xbf, Wvb, Vbf, M, DD, DD);

    // 3) V -> (B,H,DK,S) so PV B-fragments are contiguous
    transpose_v<<<nX / 256, 256, 0, stream>>>(Vbf, Vtb);

    // 4) causal flash attention (TDM-staged K/V)
    dim3 agrid(SS / 64, BB * HH);
    attn_kernel<<<agrid, 128, 0, stream>>>(Qbf, Kbf, Vtb, Attb);

    // 5) output projection, fp32 result
    gemm_nt<true><<<ggrid, 256, 0, stream>>>(Attb, Wob, (void*)Out, M, DD, DD);
}
